// UncertaintyEstimatorIterativeMambaErrorEstimation_14164802142701
// MI455X (gfx1250) — matrix operand order repaired and checked by reading, not compiled, not run on hardware
//
#include <hip/hip_runtime.h>
#include <hip/hip_bf16.h>

// Problem constants (from reference)
#define BB     2
#define TT     4
#define CC     96
#define IMG_H  48
#define IMG_W  48
#define LL     (IMG_H*IMG_W)   // 2304
#define DI     192             // d_inner
#define E2     384             // 2*d_inner
#define KD     4               // scan directions
#define NS     8               // d_state
#define RR     6               // dt_rank
#define XR     22              // x_proj rows = RR + 2*NS
#define XRP    32              // padded x_proj rows / padded dt K

typedef __attribute__((ext_vector_type(16))) __bf16 v16bf;
typedef __attribute__((ext_vector_type(8)))  float  v8f;

// ---------------------------------------------------------------------------
// WMMA GEMM:  Y[m,n] = sum_k W[m,k] * X[k,n]  (+bias[m], +activation)
// Compile-time M (mult of 16), K (mult of 32); N == LL (mult of 64).
// fp32 in memory, bf16 WMMA operands, fp32 accumulate. No masks -> no
// divergent control flow; A fragment loaded as 4x b128 per k-step.
// MODE: 0 none, 2 bias+softplus, 3 bias+leaky_relu(0.01), 4 bias+sigmoid
// ---------------------------------------------------------------------------
template<int MODE, int M, int K>
__global__ void k_gemm(const float* __restrict__ Wm, const float* __restrict__ Xm,
                       const float* __restrict__ bias, float* __restrict__ Ym,
                       long wStride, int wMod, long xStride, long yStride, long bStride)
{
    const int wave = threadIdx.x >> 5;
    const int lane = threadIdx.x & 31;
    const int half = lane >> 4;      // 0: lanes 0-15, 1: lanes 16-31
    const int lan  = lane & 15;
    const int n0   = (blockIdx.x * (blockDim.x >> 5) + wave) * 16;
    const int m0   = blockIdx.y * 16;
    const int bz   = blockIdx.z;

    const float* Wp = Wm + (long)(bz % wMod) * wStride;
    const float* Xp = Xm + (long)bz * xStride;
    float*       Yp = Ym + (long)bz * yStride;

    // Per-lane A row pointer: row = m0+lan, K-offset half*16 (contiguous k order, B matches:
    // lane half h, element j -> K=k0+16h+j).
    const float* wrow = Wp + (long)(m0 + lan) * K + half * 16;
    // Per-lane B column pointer: col = n0+lan, K = k0 + 16*half + j.
    const float* xcol = Xp + n0 + lan;

    v8f acc = {};
    #pragma unroll
    for (int k0 = 0; k0 < K; k0 += 32) {
        const float4 a0 = *(const float4*)(wrow + k0);
        const float4 a1 = *(const float4*)(wrow + k0 + 4);
        const float4 a2 = *(const float4*)(wrow + k0 + 8);
        const float4 a3 = *(const float4*)(wrow + k0 + 12);
        v16bf a;
        a[0]=(__bf16)a0.x; a[1]=(__bf16)a0.y; a[2]=(__bf16)a0.z; a[3]=(__bf16)a0.w;
        a[4]=(__bf16)a1.x; a[5]=(__bf16)a1.y; a[6]=(__bf16)a1.z; a[7]=(__bf16)a1.w;
        a[8]=(__bf16)a2.x; a[9]=(__bf16)a2.y; a[10]=(__bf16)a2.z; a[11]=(__bf16)a2.w;
        a[12]=(__bf16)a3.x; a[13]=(__bf16)a3.y; a[14]=(__bf16)a3.z; a[15]=(__bf16)a3.w;
        v16bf bfr;
        #pragma unroll
        for (int j = 0; j < 16; ++j)
            bfr[j] = (__bf16)xcol[(long)(k0 + half * 16 + j) * LL];
        acc = __builtin_amdgcn_wmma_f32_16x16x32_bf16(
            /*neg_a=*/false, a, /*neg_b=*/false, bfr,
            /*c_mod=*/(short)0, acc, /*reuse_a=*/false, /*reuse_b=*/false);
    }
    #pragma unroll
    for (int i = 0; i < 8; ++i) {
        const int row = m0 + i + half * 8;
        float v = acc[i];
        if (MODE >= 1) v += bias[(long)(bz % wMod) * bStride + row];
        if (MODE == 2) v = (v > 20.f) ? v : log1pf(__expf(v));       // softplus
        if (MODE == 3) v = (v > 0.f) ? v : 0.01f * v;                 // leaky relu
        if (MODE == 4) v = 1.f / (1.f + __expf(-v));                  // sigmoid
        Yp[(long)row * LL + n0 + lan] = v;
    }
}

// ---------------------------------------------------------------------------
// Weight padding (run once per launch): make every GEMM shape mask-free.
// ---------------------------------------------------------------------------
__global__ void k_pad_xproj_w(const float* __restrict__ src, float* __restrict__ dst)
{   // (KD,22,192) -> (KD,32,192), zero rows 22..31
    int i = blockIdx.x * 256 + threadIdx.x;
    if (i >= KD * XRP * DI) return;
    int k = i / (XRP * DI);
    int r = (i / DI) % XRP;
    int c = i % DI;
    dst[i] = (r < XR) ? src[((long)k * XR + r) * DI + c] : 0.f;
}

__global__ void k_pad_dt_w(const float* __restrict__ src, float* __restrict__ dst)
{   // (KD,192,6) -> (KD,192,32), zero cols 6..31
    int i = blockIdx.x * 256 + threadIdx.x;
    if (i >= KD * DI * XRP) return;
    int k = i / (DI * XRP);
    int d = (i / XRP) % DI;
    int r = i % XRP;
    dst[i] = (r < RR) ? src[((long)k * DI + d) * RR + r] : 0.f;
}

// ---------------------------------------------------------------------------
// Elementwise kernels
// ---------------------------------------------------------------------------
__global__ void k_init(const float* __restrict__ dz, const float* __restrict__ sg,
                       float* __restrict__ cur, float* __restrict__ prev, int n)
{
    int i = blockIdx.x * 256 + threadIdx.x;
    if (i < n) { prev[i] = dz[i]; cur[i] = dz[i] * __expf(sg[i]); }
}

__global__ void k_kl(const float* __restrict__ cur, const float* __restrict__ prev,
                     float* __restrict__ out, int n)
{
    int i = blockIdx.x * 256 + threadIdx.x;
    if (i >= n) return;
    float c = cur[i], p = prev[i];
    float v1 = c * c + 1e-6f, v2 = p * p + 1e-6f;
    out[i] = 0.5f * (v2 / v1 + __logf(v1 / v2) - 1.f);
}

// depthwise 3x3 SAME conv + bias + SiLU on xc (rows 0..DI-1 of XZ),
// writes directly into xs direction k=0 slot.
__global__ void k_dwconv_silu(const float* __restrict__ XZ, const float* __restrict__ cw,
                              const float* __restrict__ cb, float* __restrict__ xs)
{
    int l = blockIdx.x * 256 + threadIdx.x;
    int d = blockIdx.y, b = blockIdx.z;
    int h = l / IMG_W, w = l % IMG_W;
    const float* src = XZ + ((long)b * E2 + d) * LL;
    const float* k9  = cw + d * 9;
    float acc = cb[d];
    #pragma unroll
    for (int dy = -1; dy <= 1; ++dy)
        #pragma unroll
        for (int dx = -1; dx <= 1; ++dx) {
            int hh = h + dy, ww = w + dx;
            if (hh >= 0 && hh < IMG_H && ww >= 0 && ww < IMG_W)
                acc += k9[(dy + 1) * 3 + (dx + 1)] * src[hh * IMG_W + ww];
        }
    float sv = acc / (1.f + __expf(-acc));
    xs[(((long)b * KD + 0) * DI + d) * LL + l] = sv;
}

// Replicate direction 0 into directions 1..3 (transpose / reversals).
__global__ void k_build_xs(float* __restrict__ xs)
{
    int l = blockIdx.x * 256 + threadIdx.x;
    int d = blockIdx.y, b = blockIdx.z;
    long base = (long)b * KD * DI * LL;
    float v = xs[base + (long)d * LL + l];
    int h = l / IMG_W, w = l % IMG_W;
    int lt = w * IMG_H + h;
    xs[base + ((long)(1 * DI) + d) * LL + lt]            = v;
    xs[base + ((long)(2 * DI) + d) * LL + (LL - 1 - l)]  = v;
    xs[base + ((long)(3 * DI) + d) * LL + (LL - 1 - lt)] = v;
}

// ---------------------------------------------------------------------------
// Selective scan: one block per (b,k), one thread per d channel (192 threads).
// n=8 states in registers; B/C state tensors staged in LDS per 16-step chunk.
// x_dbl buffer has XRP=32 row stride; B rows at RR, C rows at RR+NS.
// ---------------------------------------------------------------------------
__global__ void k_scan(const float* __restrict__ dt, const float* __restrict__ xs,
                       const float* __restrict__ xdbl, const float* __restrict__ A_log,
                       const float* __restrict__ Dp, float* __restrict__ y)
{
    const int CHK = 16;
    __shared__ float sB[NS][CHK], sC[NS][CHK];
    int bk = blockIdx.x;
    int k  = bk % KD;
    int d  = threadIdx.x;
    const float* dtp = dt + ((long)bk * DI + d) * LL;
    const float* xp  = xs + ((long)bk * DI + d) * LL;
    const float* Bp  = xdbl + (long)bk * XRP * LL + (long)RR * LL;
    const float* Cp  = xdbl + (long)bk * XRP * LL + (long)(RR + NS) * LL;
    float* yp = y + ((long)bk * DI + d) * LL;

    float Av[NS], h[NS];
    #pragma unroll
    for (int n = 0; n < NS; ++n) {
        Av[n] = -__expf(A_log[((long)k * DI + d) * NS + n]);
        h[n]  = 0.f;
    }
    float Dv = Dp[k * DI + d];
    bool rev = (k >= 2);
    int  t   = threadIdx.x;

    for (int cc = 0; cc < LL / CHK; ++cc) {
        int lo = rev ? (LL - CHK - cc * CHK) : cc * CHK;
        __syncthreads();
        if (t < NS * CHK)
            sB[t >> 4][t & 15] = Bp[(long)(t >> 4) * LL + lo + (t & 15)];
        if (t >= 64 && t < 64 + NS * CHK) {
            int u = t - 64;
            sC[u >> 4][u & 15] = Cp[(long)(u >> 4) * LL + lo + (u & 15)];
        }
        __syncthreads();
        float dts[CHK], xv[CHK], yo[CHK];
        #pragma unroll
        for (int i = 0; i < CHK; ++i) { dts[i] = dtp[lo + i]; xv[i] = xp[lo + i]; }
        #pragma unroll
        for (int i = 0; i < CHK; ++i) {
            int ii = rev ? (CHK - 1 - i) : i;
            float dtv = dts[ii], xi = xv[ii];
            float dx  = dtv * xi;
            float acc = 0.f;
            #pragma unroll
            for (int n = 0; n < NS; ++n) {
                h[n] = h[n] * __expf(dtv * Av[n]) + dx * sB[n][ii];
                acc += h[n] * sC[n][ii];
            }
            yo[ii] = acc + Dv * xi;
        }
        #pragma unroll
        for (int i = 0; i < CHK; ++i) yp[lo + i] = yo[i];
    }
}

// ---------------------------------------------------------------------------
// Merge 4 directions + LayerNorm(d=192) + SiLU(z) gate. One wave per pixel,
// 6 channels per lane, wave32 shuffle reductions for mean/var.
// ---------------------------------------------------------------------------
__global__ void k_merge_ln_gate(const float* __restrict__ yf, const float* __restrict__ XZ,
                                const float* __restrict__ ln_g, const float* __restrict__ ln_b,
                                float* __restrict__ yact)
{
    int waveId = threadIdx.x >> 5;
    int lane   = threadIdx.x & 31;
    int b      = blockIdx.y;
    int l      = blockIdx.x * 8 + waveId;
    int h = l / IMG_W, w = l % IMG_W;
    int l1 = w * IMG_H + h;
    int l2 = LL - 1 - l;
    int l3 = LL - 1 - l1;
    const float* y0 = yf + ((long)(b * KD + 0) * DI) * LL;
    const float* y1 = yf + ((long)(b * KD + 1) * DI) * LL;
    const float* y2 = yf + ((long)(b * KD + 2) * DI) * LL;
    const float* y3 = yf + ((long)(b * KD + 3) * DI) * LL;

    float ym[6], s = 0.f, s2 = 0.f;
    #pragma unroll
    for (int q = 0; q < 6; ++q) {
        int d = lane + 32 * q;
        float v = y0[(long)d * LL + l]  + y1[(long)d * LL + l1]
                + y2[(long)d * LL + l2] + y3[(long)d * LL + l3];
        ym[q] = v; s += v; s2 += v * v;
    }
    #pragma unroll
    for (int off = 16; off; off >>= 1) {
        s  += __shfl_xor(s,  off, 32);
        s2 += __shfl_xor(s2, off, 32);
    }
    float mean = s * (1.f / DI);
    float var  = s2 * (1.f / DI) - mean * mean;
    float inv  = rsqrtf(var + 1e-5f);
    #pragma unroll
    for (int q = 0; q < 6; ++q) {
        int d = lane + 32 * q;
        float ln = (ym[q] - mean) * inv * ln_g[d] + ln_b[d];
        float z  = XZ[((long)b * E2 + DI + d) * LL + l];
        float sz = z / (1.f + __expf(-z));
        yact[((long)b * DI + d) * LL + l] = ln * sz;
    }
}

// ---------------------------------------------------------------------------
// nxt = cur*exp(s)+b ; also write concat([nxt, image]) for the merger.
// ---------------------------------------------------------------------------
__global__ void k_nxt(const float* __restrict__ cur, const float* __restrict__ s,
                      const float* __restrict__ bo, const float* __restrict__ img,
                      long imgStride, float* __restrict__ nxt, float* __restrict__ uin)
{
    int i = blockIdx.x * 256 + threadIdx.x;
    if (i >= BB * CC * LL) return;
    int b = i / (CC * LL);
    int rem = i - b * CC * LL;
    float nv = cur[i] * __expf(s[i]) + bo[i];
    nxt[i] = nv;
    uin[(long)b * DI * LL + rem]           = nv;
    uin[(long)b * DI * LL + CC * LL + rem] = img[(long)b * imgStride + rem];
}

// GroupNorm stats: one block per (b,group); groups are contiguous channel halves.
__global__ void k_gn_stats(const float* __restrict__ uin, float* __restrict__ stats)
{
    __shared__ float ss[256], sq[256];
    int bg = blockIdx.x;
    const float* base = uin + (long)bg * (DI / 2) * LL;
    float s = 0.f, s2 = 0.f;
    for (int i = threadIdx.x; i < (DI / 2) * LL; i += 256) {
        float v = base[i]; s += v; s2 += v * v;
    }
    ss[threadIdx.x] = s; sq[threadIdx.x] = s2; __syncthreads();
    for (int off = 128; off; off >>= 1) {
        if (threadIdx.x < off) {
            ss[threadIdx.x] += ss[threadIdx.x + off];
            sq[threadIdx.x] += sq[threadIdx.x + off];
        }
        __syncthreads();
    }
    if (threadIdx.x == 0) {
        float inv = 1.f / ((DI / 2) * LL);
        float m = ss[0] * inv;
        float v = sq[0] * inv - m * m;
        stats[bg * 2]     = m;
        stats[bg * 2 + 1] = rsqrtf(v + 1e-6f);
    }
}

__global__ void k_gn_apply(float* __restrict__ u, const float* __restrict__ stats,
                           const float* __restrict__ g, const float* __restrict__ bp)
{
    int i = blockIdx.x * 256 + threadIdx.x;
    if (i >= BB * DI * LL) return;
    int c  = (i / LL) % DI;
    int bg = i / ((DI / 2) * LL);
    float m = stats[bg * 2], is = stats[bg * 2 + 1];
    u[i] = (u[i] - m) * is * g[c] + bp[c];
}

__global__ void k_blend(const float* __restrict__ nxt, const float* __restrict__ u,
                        const float* __restrict__ rr, float* __restrict__ outp, long outStride)
{
    int i = blockIdx.x * 256 + threadIdx.x;
    if (i >= BB * CC * LL) return;
    int b = i / (CC * LL);
    int rem = i - b * CC * LL;
    float r = rr[0];
    outp[(long)b * outStride + rem] = r * nxt[i] + (1.f - r) * u[i];
}

// ---------------------------------------------------------------------------
// Host orchestration
// ---------------------------------------------------------------------------
struct SS2DP {
    const float *A_log, *D, *conv_b, *conv_w, *dt_proj_b, *dt_proj_w,
                *in_proj_w, *ln_b, *ln_g, *out_proj_w, *x_proj_w;
};

static void load_ss2d(SS2DP& p, void* const* d_in, int base)
{
    p.A_log      = (const float*)d_in[base + 0];
    p.D          = (const float*)d_in[base + 1];
    p.conv_b     = (const float*)d_in[base + 2];
    p.conv_w     = (const float*)d_in[base + 3];
    p.dt_proj_b  = (const float*)d_in[base + 4];
    p.dt_proj_w  = (const float*)d_in[base + 5];
    p.in_proj_w  = (const float*)d_in[base + 6];
    p.ln_b       = (const float*)d_in[base + 7];
    p.ln_g       = (const float*)d_in[base + 8];
    p.out_proj_w = (const float*)d_in[base + 9];
    p.x_proj_w   = (const float*)d_in[base + 10];
}

static void run_ss2d(const SS2DP& P, const float* pxw, const float* pdw,
                     const float* xin, long xinStride, float* outp,
                     float* XZ, float* xsb, float* xdbl, float* dtb, float* yf,
                     float* yact, hipStream_t st)
{
    // in_proj: [E2 x CC] * [CC x LL]
    k_gemm<0, E2, CC><<<dim3(LL / 64, E2 / 16, BB), 128, 0, st>>>(
        P.in_proj_w, xin, nullptr, XZ, 0, 1, xinStride, (long)E2 * LL, 0);
    k_dwconv_silu<<<dim3(LL / 256, DI, BB), 256, 0, st>>>(XZ, P.conv_w, P.conv_b, xsb);
    k_build_xs<<<dim3(LL / 256, DI, BB), 256, 0, st>>>(xsb);
    // x_proj (padded to 32 rows, per-direction weights): [32 x DI] * [DI x LL]
    k_gemm<0, XRP, DI><<<dim3(LL / 64, XRP / 16, BB * KD), 128, 0, st>>>(
        pxw, xsb, nullptr, xdbl, (long)XRP * DI, KD, (long)DI * LL, (long)XRP * LL, 0);
    // dt_proj + softplus (K padded to 32): [DI x 32] * [32 x LL]
    k_gemm<2, DI, XRP><<<dim3(LL / 64, DI / 16, BB * KD), 128, 0, st>>>(
        pdw, xdbl, P.dt_proj_b, dtb, (long)DI * XRP, KD, (long)XRP * LL,
        (long)DI * LL, DI);
    k_scan<<<dim3(BB * KD), 192, 0, st>>>(dtb, xsb, xdbl, P.A_log, P.D, yf);
    k_merge_ln_gate<<<dim3(LL / 8, BB), 256, 0, st>>>(yf, XZ, P.ln_g, P.ln_b, yact);
    // out_proj: [CC x DI] * [DI x LL]
    k_gemm<0, CC, DI><<<dim3(LL / 64, CC / 16, BB), 128, 0, st>>>(
        P.out_proj_w, yact, nullptr, outp, 0, 1, (long)DI * LL, (long)CC * LL, 0);
}

extern "C" void kernel_launch(void* const* d_in, const int* in_sizes, int n_in,
                              void* d_out, int out_size, void* d_ws, size_t ws_size,
                              hipStream_t stream)
{
    (void)in_sizes; (void)n_in; (void)out_size; (void)ws_size;
    const float* image = (const float*)d_in[0];   // (B,T,C,H,W)
    const float* dz    = (const float*)d_in[1];   // (B,C,H,W)
    const float* sigma = (const float*)d_in[2];   // (B,C,H,W)
    SS2DP Pb, Ps;
    load_ss2d(Pb, d_in, 3);   // params['mamba_bias']  (sorted leaf order)
    load_ss2d(Ps, d_in, 14);  // params['mamba_sigma']
    const float* b1       = (const float*)d_in[25];
    const float* b2       = (const float*)d_in[26];
    const float* gn_b     = (const float*)d_in[27];
    const float* gn_g     = (const float*)d_in[28];
    const float* res_rate = (const float*)d_in[29];
    const float* w1       = (const float*)d_in[30];
    const float* w2       = (const float*)d_in[31];

    float* wsf = (float*)d_ws;
    const long n1 = (long)BB * CC * LL;       // 442368 floats
    float* bufA  = wsf;                       // cur/prev/nxt rotation buffers
    float* bufB  = wsf + 1 * n1;
    float* bufC  = wsf + 2 * n1;
    float* klb   = wsf + 3 * n1;
    float* sout  = wsf + 4 * n1;
    float* bout  = wsf + 5 * n1;
    float* XZ    = wsf + 6 * n1;              // B*E2*LL  = 4*n1
    float* xsb   = wsf + 10 * n1;             // B*KD*DI*LL = 8*n1
    float* xdbl  = wsf + 18 * n1;             // B*KD*XRP*LL = 4/3*n1 in a 2*n1 slot (padded rows)
    float* dtb   = wsf + 20 * n1;             // 8*n1
    float* yf    = wsf + 28 * n1;             // 8*n1
    float* yact  = wsf + 36 * n1;             // 2*n1
    float* uin   = wsf + 38 * n1;             // 2*n1
    float* h1    = wsf + 40 * n1;             // n1
    float* uu    = wsf + 41 * n1;             // n1
    float* stats = wsf + 42 * n1;             // 8 floats
    float* pxw_s = wsf + 42 * n1 + 64;                 // KD*XRP*DI
    float* pdw_s = pxw_s + (long)KD * XRP * DI;        // KD*DI*XRP
    float* pxw_b = pdw_s + (long)KD * DI * XRP;
    float* pdw_b = pxw_b + (long)KD * XRP * DI;
    float* out   = (float*)d_out;

    // Pad odd-shaped weights once (zero-fill) so GEMMs are mask-free.
    const int pb = (KD * XRP * DI + 255) / 256;
    k_pad_xproj_w<<<pb, 256, 0, stream>>>(Ps.x_proj_w, pxw_s);
    k_pad_dt_w  <<<pb, 256, 0, stream>>>(Ps.dt_proj_w, pdw_s);
    k_pad_xproj_w<<<pb, 256, 0, stream>>>(Pb.x_proj_w, pxw_b);
    k_pad_dt_w  <<<pb, 256, 0, stream>>>(Pb.dt_proj_w, pdw_b);

    const int nElem = (int)n1;
    const int eb = (nElem + 255) / 256;
    k_init<<<eb, 256, 0, stream>>>(dz, sigma, bufA, bufB, nElem);

    float* cur = bufA; float* prev = bufB; float* nxt = bufC;
    for (int t = 0; t < TT; ++t) {
        k_kl<<<eb, 256, 0, stream>>>(cur, prev, klb, nElem);
        run_ss2d(Ps, pxw_s, pdw_s, klb, (long)CC * LL, sout,
                 XZ, xsb, xdbl, dtb, yf, yact, stream);
        run_ss2d(Pb, pxw_b, pdw_b, image + (long)t * CC * LL, (long)TT * CC * LL, bout,
                 XZ, xsb, xdbl, dtb, yf, yact, stream);
        k_nxt<<<eb, 256, 0, stream>>>(cur, sout, bout, image + (long)t * CC * LL,
                                      (long)TT * CC * LL, nxt, uin);
        k_gn_stats<<<BB * 2, 256, 0, stream>>>(uin, stats);
        const int gb = (BB * DI * LL + 255) / 256;
        k_gn_apply<<<gb, 256, 0, stream>>>(uin, stats, gn_g, gn_b);
        k_gemm<3, CC, DI><<<dim3(LL / 64, CC / 16, BB), 128, 0, stream>>>(
            w1, uin, b1, h1, 0, 1, (long)DI * LL, (long)CC * LL, 0);
        k_gemm<4, CC, CC><<<dim3(LL / 64, CC / 16, BB), 128, 0, stream>>>(
            w2, h1, b2, uu, 0, 1, (long)CC * LL, (long)CC * LL, 0);
        k_blend<<<eb, 256, 0, stream>>>(nxt, uu, res_rate,
                                        out + (long)t * CC * LL, (long)TT * CC * LL);
        float* tm = prev; prev = cur; cur = nxt; nxt = tm;
    }
}
